// EncoderBlock_75222057222872
// MI455X (gfx1250) — matrix operand order repaired and checked by reading, not compiled, not run on hardware
//
#include <hip/hip_runtime.h>
#include <hip/hip_bf16.h>
#include <math.h>

// ---------------- problem constants ----------------
#define D_MODEL 1024
#define N_HEADS 16
#define D_FF    4096
#define B_SZ    4
#define T_SEQ   2048
#define D_HEAD  64
#define M_ROWS  (B_SZ * T_SEQ)   // 8192 tokens

typedef __bf16 bf16_t;
typedef __attribute__((ext_vector_type(16))) __bf16 v16bf;
typedef __attribute__((ext_vector_type(8)))  float  v8f;

// D = A(16x32 bf16) * B(32x16 bf16) + C(16x16 f32)
__device__ __forceinline__ v8f wmma_bf16(v16bf a, v16bf b, v8f c) {
  return __builtin_amdgcn_wmma_f32_16x16x32_bf16(
      /*neg_a=*/false, a, /*neg_b=*/false, b,
      /*c_mod=*/(short)0, c, /*reuse_a=*/false, /*reuse_b=*/false);
}

// A-fragment (16x32, 16-bit): lane(hi=lane>>4) holds row (lane&15),
// K chunks {hi*8..hi*8+7} and {hi*8+16..hi*8+23}  (ISA 7.12.2)
__device__ __forceinline__ v16bf ldfragA(const bf16_t* rowp, int kc) {
  v16bf r;
  ((uint4*)&r)[0] = *(const uint4*)(rowp + kc);
  ((uint4*)&r)[1] = *(const uint4*)(rowp + kc + 16);
  return r;
}
// B-fragment (32x16, 16-bit): lane holds column N = (lane&15),
// K chunks {hi*8..hi*8+7} and {hi*8+16..hi*8+23} (stride ld)
__device__ __forceinline__ v16bf ldfragB(const bf16_t* p, int ld, int hi) {
  v16bf r;
#pragma unroll
  for (int s = 0; s < 8; ++s) {
    r[s]     = p[(size_t)(hi * 8 + s) * ld];
    r[8 + s] = p[(size_t)(16 + hi * 8 + s) * ld];
  }
  return r;
}

// -------- cross-lane butterfly reductions within each 16-lane half --------
// Prefer VALU v_permlane16_b32 (co-executes with XDL/WMMA); DS-free.
#if __has_builtin(__builtin_amdgcn_permlane16)
__device__ __forceinline__ float plx(float v, unsigned s1, unsigned s2) {
  unsigned i = __builtin_bit_cast(unsigned, v);
  i = __builtin_amdgcn_permlane16(i, i, s1, s2, false, false);
  return __builtin_bit_cast(float, i);
}
__device__ __forceinline__ float redmax16(float v) {
  v = fmaxf(v, plx(v, 0x67452301u, 0xEFCDAB89u));  // xor 1
  v = fmaxf(v, plx(v, 0x54761032u, 0xDCFE98BAu));  // xor 2
  v = fmaxf(v, plx(v, 0x32107654u, 0xBA98FEDCu));  // xor 4
  v = fmaxf(v, plx(v, 0xFEDCBA98u, 0x76543210u));  // xor 8
  return v;
}
__device__ __forceinline__ float redsum16(float v) {
  v += plx(v, 0x67452301u, 0xEFCDAB89u);
  v += plx(v, 0x54761032u, 0xDCFE98BAu);
  v += plx(v, 0x32107654u, 0xBA98FEDCu);
  v += plx(v, 0xFEDCBA98u, 0x76543210u);
  return v;
}
#else
__device__ __forceinline__ float redmax16(float v) {
  for (int mk = 1; mk < 16; mk <<= 1) v = fmaxf(v, __shfl_xor(v, mk, 32));
  return v;
}
__device__ __forceinline__ float redsum16(float v) {
  for (int mk = 1; mk < 16; mk <<= 1) v += __shfl_xor(v, mk, 32);
  return v;
}
#endif

// ---------------- f32 [N,K] -> bf16 transposed [K,N] ----------------
// One-time weight prep so GEMM B-fragments are contiguous 32B global reads.
__global__ __launch_bounds__(256) void cvt_transpose_bf16(
    const float* __restrict__ in, bf16_t* __restrict__ out, int N, int K) {
  __shared__ float tile[32][33];
  const int n0 = blockIdx.x * 32, k0 = blockIdx.y * 32;
  const int tx = threadIdx.x & 31, ty = threadIdx.x >> 5;  // 32 x 8
#pragma unroll
  for (int r = 0; r < 32; r += 8)
    tile[ty + r][tx] = in[(size_t)(n0 + ty + r) * K + k0 + tx];
  __syncthreads();
#pragma unroll
  for (int r = 0; r < 32; r += 8)
    out[(size_t)(k0 + ty + r) * N + n0 + tx] = (bf16_t)tile[tx][ty + r];
}

// ---------------- LayerNorm -> bf16 ----------------
__global__ __launch_bounds__(256) void ln_bf16_kernel(
    const float* __restrict__ x, const float* __restrict__ g,
    const float* __restrict__ beta, bf16_t* __restrict__ out) {
  const int row = blockIdx.x;
  const float* xr = x + (size_t)row * D_MODEL;
  bf16_t* orow = out + (size_t)row * D_MODEL;
  const int tid = threadIdx.x;
  float v[4];
  float s = 0.f;
#pragma unroll
  for (int j = 0; j < 4; ++j) { v[j] = xr[tid + j * 256]; s += v[j]; }
  __shared__ float red[256];
  red[tid] = s; __syncthreads();
  for (int off = 128; off > 0; off >>= 1) {
    if (tid < off) red[tid] += red[tid + off];
    __syncthreads();
  }
  const float mu = red[0] * (1.f / D_MODEL);
  __syncthreads();
  float q = 0.f;
#pragma unroll
  for (int j = 0; j < 4; ++j) { float d = v[j] - mu; q += d * d; }
  red[tid] = q; __syncthreads();
  for (int off = 128; off > 0; off >>= 1) {
    if (tid < off) red[tid] += red[tid + off];
    __syncthreads();
  }
  const float rstd = rsqrtf(red[0] * (1.f / D_MODEL) + 1e-5f);
#pragma unroll
  for (int j = 0; j < 4; ++j) {
    const int i = tid + j * 256;
    orow[i] = (bf16_t)((v[j] - mu) * rstd * g[i] + beta[i]);
  }
}

// ---- LDS-free tiled WMMA GEMM: out = A[M,K] * W^T + bias (Wt = [K,N]) ----
// Block: 128 threads / 4 waves; per-wave tile 16(M) x 128(N); BK = 32.
enum { EP_QP = 0, EP_KT = 1, EP_VP = 2, EP_RESID_F32 = 3, EP_GELU_BF16 = 4 };

template <int MODE>
__global__ __launch_bounds__(128) void gemm_wmma_kernel(
    const bf16_t* __restrict__ A,   // [M,K] bf16 row-major
    const bf16_t* __restrict__ Wt,  // [K,N] bf16 (pre-transposed weights)
    const float* __restrict__ bias, const float* __restrict__ resid,
    float* __restrict__ outf, bf16_t* __restrict__ outb, int N, int K) {
  const int m0 = blockIdx.y * 64;
  const int n0 = blockIdx.x * 128;
  const int tid = threadIdx.x;
  const int wave = tid >> 5, lane = tid & 31;
  const int l16 = lane & 15, hi = lane >> 4;
  const int kl = l16 + hi * 16;   // B-frag row within the 32-wide k-slab

  const v8f vzero = {0.f, 0.f, 0.f, 0.f, 0.f, 0.f, 0.f, 0.f};
  v8f acc[8];
#pragma unroll
  for (int i = 0; i < 8; ++i) acc[i] = vzero;

  const bf16_t* arow = A + (size_t)(m0 + wave * 16 + l16) * K;   // per-lane A row
  const bf16_t* bptr = Wt + n0 + l16;                            // per-lane B column

  for (int k0 = 0; k0 < K; k0 += 32) {
    const v16bf afr = ldfragA(arow + k0, hi * 8);
    if (k0 + 32 < K) __builtin_prefetch(arow + k0 + 32, 0, 1);   // global_prefetch
#pragma unroll
    for (int nt = 0; nt < 8; ++nt) {
      const v16bf bfr = ldfragB(bptr + nt * 16, N, hi);
      acc[nt] = wmma_bf16(afr, bfr, acc[nt]);
    }
    bptr += (size_t)32 * N;
  }

  // epilogue: C layout lane holds [M = r + 8*hi][N = l16]
#pragma unroll
  for (int nt = 0; nt < 8; ++nt) {
    const float* af = (const float*)&acc[nt];
#pragma unroll
    for (int r = 0; r < 8; ++r) {
      const int m = m0 + wave * 16 + r + 8 * hi;
      const int n = n0 + nt * 16 + l16;
      float v = af[r] + bias[n];
      if (MODE == EP_RESID_F32) {
        outf[(size_t)m * N + n] = v + resid[(size_t)m * N + n];
      } else if (MODE == EP_GELU_BF16) {
        const float ge = 0.5f * v * (1.0f + erff(v * 0.70710678118654752f));
        outb[(size_t)m * N + n] = (bf16_t)ge;
      } else {
        const int bi = m >> 11, t = m & (T_SEQ - 1);   // T = 2048
        const int h = n >> 6, d = n & (D_HEAD - 1);
        if (MODE == EP_QP)
          outb[(((size_t)bi * N_HEADS + h) * T_SEQ + t) * D_HEAD + d] = (bf16_t)(v * 0.125f);
        else if (MODE == EP_KT)
          outb[(((size_t)bi * N_HEADS + h) * D_HEAD + d) * T_SEQ + t] = (bf16_t)v;
        else  // EP_VP
          outb[(((size_t)bi * N_HEADS + h) * T_SEQ + t) * D_HEAD + d] = (bf16_t)v;
      }
    }
  }
}

// ---------------- fused flash attention (one wave = 16 queries) ----------------
__global__ __launch_bounds__(128) void attn_kernel(
    const bf16_t* __restrict__ qp,   // [B,H,T,64] bf16, pre-scaled by 1/8
    const bf16_t* __restrict__ kt,   // [B,H,64,T] bf16
    const bf16_t* __restrict__ vp,   // [B,H,T,64] bf16
    bf16_t* __restrict__ obuf) {     // [B*T, D_MODEL] bf16
  const int bh = blockIdx.y;
  const int bi = bh >> 4, h = bh & (N_HEADS - 1);
  const int tid = threadIdx.x;
  const int wave = tid >> 5, lane = tid & 31;
  const int l16 = lane & 15, hi = lane >> 4;
  const int q0 = blockIdx.x * 64 + wave * 16;

  const bf16_t* qph = qp + (size_t)bh * T_SEQ * D_HEAD;
  const bf16_t* kth = kt + (size_t)bh * D_HEAD * T_SEQ;
  const bf16_t* vph = vp + (size_t)bh * T_SEQ * D_HEAD;

  // Q fragments for K-dim 0..31 and 32..63 (loaded once, reused 64 times)
  const v16bf aq0 = ldfragA(qph + (size_t)(q0 + l16) * D_HEAD + 0,  hi * 8);
  const v16bf aq1 = ldfragA(qph + (size_t)(q0 + l16) * D_HEAD + 32, hi * 8);

  const v8f vzero = {0.f, 0.f, 0.f, 0.f, 0.f, 0.f, 0.f, 0.f};
  v8f accO[4];
#pragma unroll
  for (int i = 0; i < 4; ++i) accO[i] = vzero;
  float mrow[8], lrow[8];
#pragma unroll
  for (int r = 0; r < 8; ++r) { mrow[r] = -1e30f; lrow[r] = 0.f; }

  // Per-wave P staging buffer (C-layout -> A-frag). Only the owning wave
  // touches its slice, and per-wave LDS ops execute in order (ISA 7.3),
  // so no block barrier is needed -- only a compiler reordering fence.
  __shared__ bf16_t pst[4][16 * 32];

  const int kl = l16 + hi * 16;
  for (int j0 = 0; j0 < T_SEQ; j0 += 32) {
    // S = Q*K^T for 32 keys -> two 16x16 f32 tiles
    v8f s0 = vzero, s1 = vzero;
    {
      const v16bf b00 = ldfragB(kth + (size_t)0 * T_SEQ + j0 + l16, T_SEQ, hi);
      const v16bf b01 = ldfragB(kth + (size_t)0 * T_SEQ + j0 + 16 + l16, T_SEQ, hi);
      s0 = wmma_bf16(aq0, b00, s0);
      s1 = wmma_bf16(aq0, b01, s1);
      const v16bf b10 = ldfragB(kth + (size_t)32 * T_SEQ + j0 + l16, T_SEQ, hi);
      const v16bf b11 = ldfragB(kth + (size_t)32 * T_SEQ + j0 + 16 + l16, T_SEQ, hi);
      s0 = wmma_bf16(aq1, b10, s0);
      s1 = wmma_bf16(aq1, b11, s1);
    }
    // online softmax update (row M = r + 8*hi lives in a 16-lane half)
    const float* s0f = (const float*)&s0;
    const float* s1f = (const float*)&s1;
    float p0[8], p1[8], sc[8];
#pragma unroll
    for (int r = 0; r < 8; ++r) {
      const float mx = redmax16(fmaxf(s0f[r], s1f[r]));
      const float mn = fmaxf(mrow[r], mx);
      sc[r] = __expf(mrow[r] - mn);
      p0[r] = __expf(s0f[r] - mn);
      p1[r] = __expf(s1f[r] - mn);
      const float rs = redsum16(p0[r] + p1[r]);
      lrow[r] = lrow[r] * sc[r] + rs;
      mrow[r] = mn;
    }
#pragma unroll
    for (int nt = 0; nt < 4; ++nt) {
      float* ao = (float*)&accO[nt];
#pragma unroll
      for (int r = 0; r < 8; ++r) ao[r] *= sc[r];
    }
    // stage P(16x32) row-major in per-wave LDS, then reload as A-fragment
#pragma unroll
    for (int r = 0; r < 8; ++r) {
      const int row = r + 8 * hi;
      pst[wave][row * 32 + l16]      = (bf16_t)p0[r];
      pst[wave][row * 32 + 16 + l16] = (bf16_t)p1[r];
    }
    asm volatile("" ::: "memory");   // keep DS store -> DS load order in IR
    const v16bf pa = ldfragA(&pst[wave][l16 * 32], hi * 8);
#pragma unroll
    for (int nt = 0; nt < 4; ++nt) {
      const v16bf bv = ldfragB(vph + (size_t)j0 * D_HEAD + nt * 16 + l16, D_HEAD, hi);
      accO[nt] = wmma_bf16(pa, bv, accO[nt]);
    }
    asm volatile("" ::: "memory");
  }
  // normalize and write to [B*T, D_MODEL] bf16 for the O projection
#pragma unroll
  for (int r = 0; r < 8; ++r) lrow[r] = 1.f / lrow[r];
#pragma unroll
  for (int nt = 0; nt < 4; ++nt) {
    const float* ao = (const float*)&accO[nt];
#pragma unroll
    for (int r = 0; r < 8; ++r) {
      const int t = q0 + r + 8 * hi;
      const int feat = h * D_HEAD + nt * 16 + l16;
      obuf[((size_t)bi * T_SEQ + t) * D_MODEL + feat] = (bf16_t)(ao[r] * lrow[r]);
    }
  }
}

// ---------------- host orchestration ----------------
extern "C" void kernel_launch(void* const* d_in, const int* in_sizes, int n_in,
                              void* d_out, int out_size, void* d_ws, size_t ws_size,
                              hipStream_t stream) {
  const float* x    = (const float*)d_in[0];
  const float* ln1g = (const float*)d_in[1];
  const float* ln1b = (const float*)d_in[2];
  const float* wq   = (const float*)d_in[3];
  const float* bq   = (const float*)d_in[4];
  const float* wk   = (const float*)d_in[5];
  const float* bk   = (const float*)d_in[6];
  const float* wv   = (const float*)d_in[7];
  const float* bv   = (const float*)d_in[8];
  const float* wo   = (const float*)d_in[9];
  const float* bo   = (const float*)d_in[10];
  const float* ln2g = (const float*)d_in[11];
  const float* ln2b = (const float*)d_in[12];
  const float* w1   = (const float*)d_in[13];
  const float* b1   = (const float*)d_in[14];
  const float* w2   = (const float*)d_in[15];
  const float* b2   = (const float*)d_in[16];
  float* out = (float*)d_out;

  // workspace carve-out (~200 MB)
  char* ws = (char*)d_ws;
  size_t off = 0;
  auto carve = [&](size_t bytes) {
    char* p = ws + off;
    off += (bytes + 255) & ~(size_t)255;
    return p;
  };
  bf16_t* wqb  = (bf16_t*)carve((size_t)D_MODEL * D_MODEL * 2);  // [K,N] transposed
  bf16_t* wkb  = (bf16_t*)carve((size_t)D_MODEL * D_MODEL * 2);
  bf16_t* wvb  = (bf16_t*)carve((size_t)D_MODEL * D_MODEL * 2);
  bf16_t* wob  = (bf16_t*)carve((size_t)D_MODEL * D_MODEL * 2);
  bf16_t* w1b  = (bf16_t*)carve((size_t)D_FF * D_MODEL * 2);     // [1024, 4096]
  bf16_t* w2b  = (bf16_t*)carve((size_t)D_MODEL * D_FF * 2);     // [4096, 1024]
  bf16_t* h1   = (bf16_t*)carve((size_t)M_ROWS * D_MODEL * 2);   // LN1 out; aliased by attn out
  bf16_t* qpb  = (bf16_t*)carve((size_t)M_ROWS * D_MODEL * 2);
  bf16_t* ktb  = (bf16_t*)carve((size_t)M_ROWS * D_MODEL * 2);
  bf16_t* vpb  = (bf16_t*)carve((size_t)M_ROWS * D_MODEL * 2);
  float*  x1   = (float*) carve((size_t)M_ROWS * D_MODEL * 4);   // x + attn-proj
  bf16_t* h2   = (bf16_t*)carve((size_t)M_ROWS * D_MODEL * 2);   // LN2 out
  bf16_t* ffn1 = (bf16_t*)carve((size_t)M_ROWS * D_FF * 2);      // gelu(h2 @ w1^T + b1)
  bf16_t* obuf = h1;  // h1 dead after the QKV GEMMs

  // one-time weight prep: f32 [N,K] -> bf16 [K,N]
  auto cvtT = [&](const float* s, bf16_t* d, int N, int K) {
    cvt_transpose_bf16<<<dim3(N / 32, K / 32), 256, 0, stream>>>(s, d, N, K);
  };
  cvtT(wq, wqb, D_MODEL, D_MODEL);
  cvtT(wk, wkb, D_MODEL, D_MODEL);
  cvtT(wv, wvb, D_MODEL, D_MODEL);
  cvtT(wo, wob, D_MODEL, D_MODEL);
  cvtT(w1, w1b, D_FF, D_MODEL);
  cvtT(w2, w2b, D_MODEL, D_FF);

  const dim3 blk(128);
  const dim3 g_d(D_MODEL / 128, M_ROWS / 64);   // (8,128)
  const dim3 g_ff(D_FF / 128, M_ROWS / 64);     // (32,128)

  // LN1 -> bf16
  ln_bf16_kernel<<<M_ROWS, 256, 0, stream>>>(x, ln1g, ln1b, h1);
  // Q/K/V projections, packed straight into attention layouts
  gemm_wmma_kernel<EP_QP><<<g_d, blk, 0, stream>>>(h1, wqb, bq, nullptr, nullptr, qpb, D_MODEL, D_MODEL);
  gemm_wmma_kernel<EP_KT><<<g_d, blk, 0, stream>>>(h1, wkb, bk, nullptr, nullptr, ktb, D_MODEL, D_MODEL);
  gemm_wmma_kernel<EP_VP><<<g_d, blk, 0, stream>>>(h1, wvb, bv, nullptr, nullptr, vpb, D_MODEL, D_MODEL);
  // fused flash attention (no score matrix in HBM)
  attn_kernel<<<dim3(T_SEQ / 64, B_SZ * N_HEADS), blk, 0, stream>>>(qpb, ktb, vpb, obuf);
  // O projection + residual -> x1 (f32)
  gemm_wmma_kernel<EP_RESID_F32><<<g_d, blk, 0, stream>>>(obuf, wob, bo, x, x1, nullptr, D_MODEL, D_MODEL);
  // LN2 -> bf16
  ln_bf16_kernel<<<M_ROWS, 256, 0, stream>>>(x1, ln2g, ln2b, h2);
  // FFN up + exact GELU -> bf16
  gemm_wmma_kernel<EP_GELU_BF16><<<g_ff, blk, 0, stream>>>(h2, w1b, b1, nullptr, nullptr, ffn1, D_FF, D_MODEL);
  // FFN down + residual -> final f32 output
  gemm_wmma_kernel<EP_RESID_F32><<<g_d, blk, 0, stream>>>(ffn1, w2b, b2, x1, out, nullptr, D_MODEL, D_FF);
}
